// SurpriseBasedMemory_30906584662253
// MI455X (gfx1250) — matrix operand order repaired and checked by reading, not compiled, not run on hardware
//
#include <hip/hip_runtime.h>
#include <hip/hip_bf16.h>

// ---------------------------------------------------------------------------
// Types / WMMA helpers (CDNA5 gfx1250, wave32)
// ---------------------------------------------------------------------------
typedef __bf16 bf16_t;
typedef __attribute__((ext_vector_type(16))) __bf16 v16bf;
typedef __attribute__((ext_vector_type(8)))  float  v8f;
typedef __attribute__((ext_vector_type(2)))  float  v2f;

union FragU { v16bf v; uint4 q[2]; };

#define MM   16384   // memory slots
#define BB   512     // batch
#define DD   256     // feature dim
#define AA   264     // attn dim
#define APAD 288     // A padded to multiple of 32 (K padding)
#define HP   512     // 8 heads * 64 padded head-dim
#define HDS  33      // real head dim

// A-matrix fragment (16x32 bf16, MxK), row-major source.
// lane l: row = l&15 ; elems j=0..15 <- K = k0 + 16*(l>=16) + j
__device__ inline v16bf load_fragA(const bf16_t* __restrict__ rowp, int k0) {
  int lane = threadIdx.x & 31;
  int half = (lane & 16) ? 16 : 0;
  FragU u;
  u.q[0] = *(const uint4*)(rowp + k0 + half);
  u.q[1] = *(const uint4*)(rowp + k0 + half + 8);
  return u.v;
}

// B-matrix fragment (32x16 bf16, KxN) built from BT (row n holds column n of B).
// lane l: col = l&15 ; elems j=0..15 <- K = k0 + 16*(l>=16) + j   (32 contiguous bytes)
__device__ inline v16bf load_fragB(const bf16_t* __restrict__ rowp, int k0) {
  int lane = threadIdx.x & 31;
  int koff = (lane & 16) ? 16 : 0;
  FragU u;
  u.q[0] = *(const uint4*)(rowp + k0 + koff);
  u.q[1] = *(const uint4*)(rowp + k0 + koff + 8);
  return u.v;
}

__device__ inline v8f wmma_bf16(v16bf a, v16bf b, v8f c) {
  return __builtin_amdgcn_wmma_f32_16x16x32_bf16(false, a, false, b, (short)0, c,
                                                 false, false);
}

// f32 fragment (16x4 / 4x16): lanes 0-15 hold K={k0,k0+1}, lanes 16-31 K={k0+2,k0+3}.
__device__ inline v2f load_frag_f32(const float* __restrict__ rowp, int k0) {
  int lane = threadIdx.x & 31;
  int koff = (lane & 16) ? 2 : 0;
  return *(const v2f*)(rowp + k0 + koff);
}

__device__ inline v8f wmma_f32(v2f a, v2f b, v8f c) {
  return __builtin_amdgcn_wmma_f32_16x16x4_f32(false, a, false, b, (short)0, c,
                                               false, false);
}

// ---------------------------------------------------------------------------
// Generic bf16 WMMA GEMM:  C[M,N] = A[M,K] * B[K,N] (+bias), B given as BT[N,K]
// One 16x16 tile per wave, 8 waves (256 threads) per block. K % 32 == 0, M % 16 == 0.
// ---------------------------------------------------------------------------
__global__ void gemm_bf16_kernel(const bf16_t* __restrict__ A, int lda,
                                 const bf16_t* __restrict__ BT, int ldb,
                                 float* __restrict__ C, int ldc,
                                 int M, int N, int K,
                                 const float* __restrict__ bias) {
  int wave = threadIdx.x >> 5;
  int lane = threadIdx.x & 31;
  int Nt = (N + 15) >> 4;
  int Mt = M >> 4;
  int t = blockIdx.x * (blockDim.x >> 5) + wave;
  if (t >= Mt * Nt) return;                 // wave-uniform
  int tm = t / Nt, tn = t % Nt;
  int ar = tm * 16 + (lane & 15);
  int br = tn * 16 + (lane & 15);
  if (br >= N) br = N - 1;                  // clamp (outputs guarded below)
  const bf16_t* arow = A  + (size_t)ar * lda;
  const bf16_t* brow = BT + (size_t)br * ldb;
  v8f acc = {};
  for (int k0 = 0; k0 < K; k0 += 32) {
    v16bf a = load_fragA(arow, k0);
    v16bf b = load_fragB(brow, k0);
    acc = wmma_bf16(a, b, acc);
  }
  int col = tn * 16 + (lane & 15);
  if (col < N) {
    float bvv = bias ? bias[col] : 0.f;
    int rbase = tm * 16 + ((lane >> 4) << 3);   // C: lane holds N=l&15, M=r+8*(l>>4)
#pragma unroll
    for (int r = 0; r < 8; ++r)
      C[(size_t)(rbase + r) * ldc + col] = acc[r] + bvv;
  }
}

// ---------------------------------------------------------------------------
// Exact-fp32 WMMA GEMM via V_WMMA_F32_16X16X4_F32 (for the distance matrix,
// which feeds min/sqrt/threshold/argmin and is precision-critical).
// C[M,N] = A[M,K] * BT[N,K]^T.  K % 4 == 0, M,N multiples of 16, no bias.
// ---------------------------------------------------------------------------
__global__ void gemm_f32_kernel(const float* __restrict__ A, int lda,
                                const float* __restrict__ BT, int ldb,
                                float* __restrict__ C, int ldc,
                                int M, int N, int K) {
  int wave = threadIdx.x >> 5;
  int lane = threadIdx.x & 31;
  int Nt = N >> 4;
  int Mt = M >> 4;
  int t = blockIdx.x * (blockDim.x >> 5) + wave;
  if (t >= Mt * Nt) return;                 // wave-uniform
  int tm = t / Nt, tn = t % Nt;
  const float* arow = A  + (size_t)(tm * 16 + (lane & 15)) * lda;
  const float* brow = BT + (size_t)(tn * 16 + (lane & 15)) * ldb;
  v8f acc = {};
  for (int k0 = 0; k0 < K; k0 += 4) {
    v2f a = load_frag_f32(arow, k0);
    v2f b = load_frag_f32(brow, k0);
    acc = wmma_f32(a, b, acc);
  }
  int col = tn * 16 + (lane & 15);
  int rbase = tm * 16 + ((lane >> 4) << 3);
#pragma unroll
  for (int r = 0; r < 8; ++r)
    C[(size_t)(rbase + r) * ldc + col] = acc[r];
}

// ---------------------------------------------------------------------------
// Small utility kernels
// ---------------------------------------------------------------------------
__global__ void cvt_pad_kernel(const float* __restrict__ src, bf16_t* __restrict__ dst,
                               int R, int C, int Cout) {
  size_t idx = (size_t)blockIdx.x * blockDim.x + threadIdx.x;
  if (idx >= (size_t)R * Cout) return;
  int c = (int)(idx % Cout);
  size_t r = idx / Cout;
  dst[idx] = (c < C) ? (bf16_t)src[r * C + c] : (bf16_t)0.f;
}

// dst[n*Kpad + k] = k<K ? src[k*srcStride + colOfs + n] : 0   (transpose + pad + cvt)
__global__ void make_bt_kernel(const float* __restrict__ src, int srcStride, int colOfs,
                               int K, int N, int Kpad, bf16_t* __restrict__ dst) {
  size_t idx = (size_t)blockIdx.x * blockDim.x + threadIdx.x;
  if (idx >= (size_t)N * Kpad) return;
  int k = (int)(idx % Kpad);
  int n = (int)(idx / Kpad);
  dst[idx] = (k < K) ? (bf16_t)src[(size_t)k * srcStride + colOfs + n] : (bf16_t)0.f;
}

__global__ void rownorm_kernel(const float* __restrict__ src, int C,
                               float* __restrict__ out, int do_sqrt) {
  __shared__ float red[256];
  int r = blockIdx.x;
  float s = 0.f;
  for (int c = threadIdx.x; c < C; c += blockDim.x) {
    float v = src[(size_t)r * C + c];
    s += v * v;
  }
  red[threadIdx.x] = s;
  __syncthreads();
  for (int o = 128; o > 0; o >>= 1) {
    if ((int)threadIdx.x < o) red[threadIdx.x] += red[threadIdx.x + o];
    __syncthreads();
  }
  if (threadIdx.x == 0) out[r] = do_sqrt ? sqrtf(red[0]) : red[0];
}

__global__ void surprise_kernel(const float* __restrict__ dot, const float* __restrict__ fn2,
                                const float* __restrict__ kn2, const float* __restrict__ gn,
                                float* __restrict__ surprise, int* __restrict__ memorize) {
  __shared__ float red[256];
  int b = blockIdx.x;
  const float* row = dot + (size_t)b * MM;
  float f = fn2[b];
  float mn = __builtin_inff();
  for (int m = threadIdx.x; m < MM; m += blockDim.x) {
    float d2 = f + kn2[m] - 2.f * row[m];
    mn = fminf(mn, d2);
  }
  red[threadIdx.x] = mn;
  __syncthreads();
  for (int o = 128; o > 0; o >>= 1) {
    if ((int)threadIdx.x < o) red[threadIdx.x] = fminf(red[threadIdx.x], red[threadIdx.x + o]);
    __syncthreads();
  }
  if (threadIdx.x == 0) {
    float md = sqrtf(fmaxf(red[0], 0.f));
    float s = gn[b] * md;
    surprise[b] = s;
    memorize[b] = (s > 0.1f) ? 1 : 0;
  }
}

// Sequential eviction scan: single 1024-thread workgroup, priority array in dynamic LDS.
__global__ void scan_kernel(const float* __restrict__ mem_ages,
                            const float* __restrict__ mem_surprise,
                            const float* __restrict__ surprise,
                            const int* __restrict__ memorize,
                            int* __restrict__ last_writer) {
  extern __shared__ unsigned char smem_raw[];
  float* prio = (float*)smem_raw;            // MM floats = 64 KB
  __shared__ float rv[32];
  __shared__ int   ri[32];
  int tid = threadIdx.x, nthr = blockDim.x;
  int lane = tid & 31, warp = tid >> 5, nwarp = nthr >> 5;
  for (int m = tid; m < MM; m += nthr) {
    prio[m] = mem_surprise[m] * __powf(0.95f, mem_ages[m]);
    last_writer[m] = -1;
  }
  __syncthreads();
  for (int b = 0; b < BB; ++b) {
    if (memorize[b]) {
      float bvv = __builtin_inff();
      int bi = MM;
      for (int m = tid; m < MM; m += nthr) {
        float v = prio[m];
        if (v < bvv) { bvv = v; bi = m; }    // ascending m -> first-min kept
      }
      for (int o = 16; o > 0; o >>= 1) {
        float ov = __shfl_xor(bvv, o, 32);
        int   oi = __shfl_xor(bi, o, 32);
        if (ov < bvv || (ov == bvv && oi < bi)) { bvv = ov; bi = oi; }
      }
      if (lane == 0) { rv[warp] = bvv; ri[warp] = bi; }
      __syncthreads();
      if (tid == 0) {
        float best = rv[0]; int besti = ri[0];
        for (int w = 1; w < nwarp; ++w)
          if (rv[w] < best || (rv[w] == best && ri[w] < besti)) { best = rv[w]; besti = ri[w]; }
        last_writer[besti] = b;
        prio[besti] = surprise[b];           // age reset to 0 -> decay^0 = 1
      }
      __syncthreads();
    }
  }
}

// Post-scan: new memory (bf16) = written slot ? fk/fv row : old memory row
__global__ void apply_writes_kernel(const int* __restrict__ last_writer,
                                    const float* __restrict__ fk, const float* __restrict__ fv,
                                    const float* __restrict__ mem_keys,
                                    const float* __restrict__ mem_values,
                                    bf16_t* __restrict__ mknb, bf16_t* __restrict__ mvnb) {
  int m = blockIdx.x;
  int d = threadIdx.x;
  int b = last_writer[m];
  size_t o = (size_t)m * DD + d;
  float kv, vv;
  if (b >= 0) { kv = fk[(size_t)b * DD + d]; vv = fv[(size_t)b * DD + d]; }
  else        { kv = mem_keys[o];            vv = mem_values[o]; }
  mknb[o] = (bf16_t)kv;
  mvnb[o] = (bf16_t)vv;
}

// f32 [R,264] -> bf16 [R,512] head-padded (col h*64+d <- h*33+d, d<33 else 0)
__global__ void remap_heads_kernel(const float* __restrict__ src, bf16_t* __restrict__ dst,
                                   int R) {
  size_t idx = (size_t)blockIdx.x * blockDim.x + threadIdx.x;
  if (idx >= (size_t)R * HP) return;
  int c = (int)(idx & (HP - 1));
  size_t r = idx >> 9;
  int h = c >> 6, d = c & 63;
  dst[idx] = (d < HDS) ? (bf16_t)src[r * AA + h * HDS + d] : (bf16_t)0.f;
}

// vh f32 [M,264] -> vhT bf16 [512][M]  (row h*64+d, col m)
__global__ void remap_headsT_kernel(const float* __restrict__ vh, bf16_t* __restrict__ vhT) {
  size_t idx = (size_t)blockIdx.x * blockDim.x + threadIdx.x;
  if (idx >= (size_t)HP * MM) return;
  int m = (int)(idx & (MM - 1));
  int row = (int)(idx >> 14);
  int h = row >> 6, d = row & 63;
  vhT[idx] = (d < HDS) ? (bf16_t)vh[(size_t)m * AA + h * HDS + d] : (bf16_t)0.f;
}

// ---------------------------------------------------------------------------
// Flash attention over M=16384 slots. One wave per (16-row b-tile, head).
// S^T = Kh(16m x 64k) * Qh^T(64 x 16b)  -> softmax state is lane-local in b.
// ctx^T = Vh^T(16d x 32m) * P^T(32m x 16b)
// grid = 32 blocks * 256 threads = 256 waves = 32 b-tiles * 8 heads.
// ---------------------------------------------------------------------------
__global__ void attn_kernel(const bf16_t* __restrict__ qhb,   // [512][512]
                            const bf16_t* __restrict__ khb,   // [16384][512]
                            const bf16_t* __restrict__ vhT,   // [512][16384]
                            float* __restrict__ ctx) {        // [512][264]
  const float SCALE = 0.17407765595569785f;  // 1/sqrt(33)
  int wave = threadIdx.x >> 5;
  int lane = threadIdx.x & 31;
  int w = blockIdx.x * 8 + wave;
  int bt = w >> 3, h = w & 7;
  int n = lane & 15;
  bool lo = (lane & 16) == 0;

  const bf16_t* qrow = qhb + (size_t)(bt * 16 + n) * HP + h * 64;
  v16bf qf0 = load_fragB(qrow, 0);
  v16bf qf1 = load_fragB(qrow, 32);
  const bf16_t* vbase = vhT + (size_t)(h * 64) * MM;

  v8f acc0 = {}, acc1 = {}, acc2 = {}, acc3 = {};
  float Mb = -__builtin_inff(), Sb = 0.f;

  for (int mb = 0; mb < MM; mb += 32) {
    const bf16_t* krow0 = khb + (size_t)(mb + n) * HP + h * 64;
    const bf16_t* krow1 = krow0 + 16 * HP;
    if (mb + 64 <= MM) __builtin_prefetch(krow0 + 32 * HP, 0, 1);
    v16bf a00 = load_fragA(krow0, 0), a01 = load_fragA(krow0, 32);
    v16bf a10 = load_fragA(krow1, 0), a11 = load_fragA(krow1, 32);
    v8f s0 = {}, s1 = {};
    s0 = wmma_bf16(a00, qf0, s0);
    s0 = wmma_bf16(a01, qf1, s0);
    s1 = wmma_bf16(a10, qf0, s1);
    s1 = wmma_bf16(a11, qf1, s1);

    float e0[8], e1[8];
    float cmax = -__builtin_inff();
#pragma unroll
    for (int r = 0; r < 8; ++r) {
      e0[r] = s0[r] * SCALE;
      e1[r] = s1[r] * SCALE;
      cmax = fmaxf(cmax, fmaxf(e0[r], e1[r]));
    }
    cmax = fmaxf(cmax, __shfl_xor(cmax, 16, 32));
    float newM = fmaxf(Mb, cmax);
    float alpha = __expf(Mb - newM);
    float csum = 0.f;
#pragma unroll
    for (int r = 0; r < 8; ++r) {
      e0[r] = __expf(e0[r] - newM);
      e1[r] = __expf(e1[r] - newM);
      csum += e0[r] + e1[r];
    }
    csum += __shfl_xor(csum, 16, 32);
    Sb = Sb * alpha + csum;
    Mb = newM;

    // Build P^T B-fragment: lanes<16 need m=0..15, lanes>=16 need m=16..31.
    FragU p;
#pragma unroll
    for (int r = 0; r < 8; ++r) {
      float t0 = __shfl_xor(e0[r], 16, 32);
      float t1 = __shfl_xor(e1[r], 16, 32);
      p.v[r]     = (bf16_t)(lo ? e0[r] : t1);
      p.v[r + 8] = (bf16_t)(lo ? t0 : e1[r]);
    }
#pragma unroll
    for (int r = 0; r < 8; ++r) {
      acc0[r] *= alpha; acc1[r] *= alpha; acc2[r] *= alpha; acc3[r] *= alpha;
    }
    v16bf vf0 = load_fragA(vbase + (size_t)(0 * 16 + n) * MM, mb);
    v16bf vf1 = load_fragA(vbase + (size_t)(1 * 16 + n) * MM, mb);
    v16bf vf2 = load_fragA(vbase + (size_t)(2 * 16 + n) * MM, mb);
    v16bf vf3 = load_fragA(vbase + (size_t)(3 * 16 + n) * MM, mb);
    acc0 = wmma_bf16(vf0, p.v, acc0);
    acc1 = wmma_bf16(vf1, p.v, acc1);
    acc2 = wmma_bf16(vf2, p.v, acc2);
    acc3 = wmma_bf16(vf3, p.v, acc3);
  }

  float inv = 1.f / Sb;
  int bglob = bt * 16 + n;
  int mofs = (lane >> 4) << 3;
#pragma unroll
  for (int r = 0; r < 8; ++r) {
    int d0 = 0  + r + mofs; if (d0 < HDS) ctx[(size_t)bglob * AA + h * HDS + d0] = acc0[r] * inv;
    int d1 = 16 + r + mofs; if (d1 < HDS) ctx[(size_t)bglob * AA + h * HDS + d1] = acc1[r] * inv;
    int d2 = 32 + r + mofs; if (d2 < HDS) ctx[(size_t)bglob * AA + h * HDS + d2] = acc2[r] * inv;
    // acc3 covers d 48..63 -> always padding, never stored
  }
}

// ---------------------------------------------------------------------------
// Host-side orchestration
// ---------------------------------------------------------------------------
static inline int cdiv_sz(size_t a, size_t b) { return (int)((a + b - 1) / b); }

extern "C" void kernel_launch(void* const* d_in, const int* in_sizes, int n_in,
                              void* d_out, int out_size, void* d_ws, size_t ws_size,
                              hipStream_t stream) {
  (void)in_sizes; (void)n_in; (void)out_size; (void)ws_size;
  const float* features     = (const float*)d_in[0];
  const float* gradients    = (const float*)d_in[1];
  const float* mem_keys     = (const float*)d_in[2];
  const float* mem_values   = (const float*)d_in[3];
  const float* mem_ages     = (const float*)d_in[4];
  const float* mem_surprise = (const float*)d_in[5];
  const float* Wk   = (const float*)d_in[6];
  const float* bk   = (const float*)d_in[7];
  const float* Wv   = (const float*)d_in[8];
  const float* bv   = (const float*)d_in[9];
  const float* Wp   = (const float*)d_in[10];
  const float* bp   = (const float*)d_in[11];
  const float* Win  = (const float*)d_in[12];
  const float* b_in = (const float*)d_in[13];
  const float* Wout = (const float*)d_in[14];
  const float* bout = (const float*)d_in[15];
  const float* Wo   = (const float*)d_in[16];
  const float* bo   = (const float*)d_in[17];
  float* out = (float*)d_out;

  char* ws = (char*)d_ws;
  size_t off = 0;
  auto alloc = [&](size_t bytes) -> char* {
    char* p = ws + off;
    off = (off + bytes + 255) & ~(size_t)255;
    return p;
  };

  bf16_t* fb     = (bf16_t*)alloc((size_t)BB * DD * 2);
  float*  dotb   = (float*) alloc((size_t)BB * MM * 4);
  float*  fn2    = (float*) alloc(BB * 4);
  float*  kn2    = (float*) alloc(MM * 4);
  float*  gn     = (float*) alloc(BB * 4);
  float*  surp   = (float*) alloc(BB * 4);
  int*    memo   = (int*)   alloc(BB * 4);
  int*    lastw  = (int*)   alloc(MM * 4);
  bf16_t* WkT    = (bf16_t*)alloc((size_t)DD * DD * 2);
  bf16_t* WvT    = (bf16_t*)alloc((size_t)DD * DD * 2);
  bf16_t* WpT    = (bf16_t*)alloc((size_t)AA * DD * 2);
  float*  fk     = (float*) alloc((size_t)BB * DD * 4);
  float*  fv     = (float*) alloc((size_t)BB * DD * 4);
  bf16_t* mknb   = (bf16_t*)alloc((size_t)MM * DD * 2);
  bf16_t* mvnb   = (bf16_t*)alloc((size_t)MM * DD * 2);
  float*  kf     = (float*) alloc((size_t)MM * AA * 4);
  float*  vf     = (float*) alloc((size_t)MM * AA * 4);
  float*  qf     = (float*) alloc((size_t)BB * AA * 4);
  bf16_t* kfb    = (bf16_t*)alloc((size_t)MM * APAD * 2);
  bf16_t* vfb    = (bf16_t*)alloc((size_t)MM * APAD * 2);
  bf16_t* qb     = (bf16_t*)alloc((size_t)BB * APAD * 2);
  bf16_t* WqTa   = (bf16_t*)alloc((size_t)AA * APAD * 2);
  bf16_t* WkTa   = (bf16_t*)alloc((size_t)AA * APAD * 2);
  bf16_t* WvTa   = (bf16_t*)alloc((size_t)AA * APAD * 2);
  float*  kh     = (float*) alloc((size_t)MM * AA * 4);
  float*  vh     = (float*) alloc((size_t)MM * AA * 4);
  float*  qh     = (float*) alloc((size_t)BB * AA * 4);
  bf16_t* khb    = (bf16_t*)alloc((size_t)MM * HP * 2);
  bf16_t* qhb    = (bf16_t*)alloc((size_t)BB * HP * 2);
  bf16_t* vhT    = (bf16_t*)alloc((size_t)HP * MM * 2);
  float*  ctx    = (float*) alloc((size_t)BB * AA * 4);
  bf16_t* ctxb   = (bf16_t*)alloc((size_t)BB * APAD * 2);
  bf16_t* WoutT  = (bf16_t*)alloc((size_t)AA * APAD * 2);
  float*  ctx2   = (float*) alloc((size_t)BB * AA * 4);
  bf16_t* ctx2b  = (bf16_t*)alloc((size_t)BB * APAD * 2);
  bf16_t* WoT    = (bf16_t*)alloc((size_t)DD * APAD * 2);

  auto gemm = [&](const bf16_t* A, int lda, const bf16_t* BT, int ldb,
                  float* C, int ldc, int M_, int N_, int K_, const float* bias) {
    int tiles = (M_ >> 4) * ((N_ + 15) >> 4);
    gemm_bf16_kernel<<<cdiv_sz(tiles, 8), 256, 0, stream>>>(A, lda, BT, ldb, C, ldc,
                                                            M_, N_, K_, bias);
  };

  // ---- Stage 1: surprise = ||grad|| * min cdist(features, mem_keys) ----
  // Exact fp32 WMMA for the distance GEMM: operands are the raw f32 inputs.
  cvt_pad_kernel<<<cdiv_sz((size_t)BB * DD, 256), 256, 0, stream>>>(features, fb, BB, DD, DD);
  rownorm_kernel<<<BB, 256, 0, stream>>>(features, DD, fn2, 0);
  rownorm_kernel<<<MM, 256, 0, stream>>>(mem_keys, DD, kn2, 0);
  rownorm_kernel<<<BB, 256, 0, stream>>>(gradients, DD, gn, 1);
  {
    int tiles = (BB >> 4) * (MM >> 4);
    gemm_f32_kernel<<<cdiv_sz(tiles, 8), 256, 0, stream>>>(features, DD, mem_keys, DD,
                                                           dotb, MM, BB, MM, DD);
  }
  surprise_kernel<<<BB, 256, 0, stream>>>(dotb, fn2, kn2, gn, surp, memo);

  // ---- Stage 2: sequential eviction scan + memory write ----
  scan_kernel<<<1, 1024, (size_t)MM * 4, stream>>>(mem_ages, mem_surprise, surp, memo, lastw);
  make_bt_kernel<<<cdiv_sz((size_t)DD * DD, 256), 256, 0, stream>>>(Wk, DD, 0, DD, DD, DD, WkT);
  make_bt_kernel<<<cdiv_sz((size_t)DD * DD, 256), 256, 0, stream>>>(Wv, DD, 0, DD, DD, DD, WvT);
  gemm(fb, DD, WkT, DD, fk, DD, BB, DD, DD, bk);
  gemm(fb, DD, WvT, DD, fv, DD, BB, DD, DD, bv);
  apply_writes_kernel<<<MM, DD, 0, stream>>>(lastw, fk, fv, mem_keys, mem_values, mknb, mvnb);

  // ---- Stage 3: projections q/kf/vf then qh/kh/vh ----
  make_bt_kernel<<<cdiv_sz((size_t)AA * DD, 256), 256, 0, stream>>>(Wp, AA, 0, DD, AA, DD, WpT);
  gemm(mknb, DD, WpT, DD, kf, AA, MM, AA, DD, bp);
  gemm(mvnb, DD, WpT, DD, vf, AA, MM, AA, DD, bp);
  gemm(fb,   DD, WpT, DD, qf, AA, BB, AA, DD, bp);
  cvt_pad_kernel<<<cdiv_sz((size_t)MM * APAD, 256), 256, 0, stream>>>(kf, kfb, MM, AA, APAD);
  cvt_pad_kernel<<<cdiv_sz((size_t)MM * APAD, 256), 256, 0, stream>>>(vf, vfb, MM, AA, APAD);
  cvt_pad_kernel<<<cdiv_sz((size_t)BB * APAD, 256), 256, 0, stream>>>(qf, qb, BB, AA, APAD);
  make_bt_kernel<<<cdiv_sz((size_t)AA * APAD, 256), 256, 0, stream>>>(Win, 3 * AA, 0,      AA, AA, APAD, WqTa);
  make_bt_kernel<<<cdiv_sz((size_t)AA * APAD, 256), 256, 0, stream>>>(Win, 3 * AA, AA,     AA, AA, APAD, WkTa);
  make_bt_kernel<<<cdiv_sz((size_t)AA * APAD, 256), 256, 0, stream>>>(Win, 3 * AA, 2 * AA, AA, AA, APAD, WvTa);
  gemm(kfb, APAD, WkTa, APAD, kh, AA, MM, AA, APAD, b_in + AA);
  gemm(vfb, APAD, WvTa, APAD, vh, AA, MM, AA, APAD, b_in + 2 * AA);
  gemm(qb,  APAD, WqTa, APAD, qh, AA, BB, AA, APAD, b_in);
  remap_heads_kernel<<<cdiv_sz((size_t)MM * HP, 256), 256, 0, stream>>>(kh, khb, MM);
  remap_heads_kernel<<<cdiv_sz((size_t)BB * HP, 256), 256, 0, stream>>>(qh, qhb, BB);
  remap_headsT_kernel<<<cdiv_sz((size_t)HP * MM, 256), 256, 0, stream>>>(vh, vhT);

  // ---- Stage 4: flash attention ----
  attn_kernel<<<32, 256, 0, stream>>>(qhb, khb, vhT, ctx);

  // ---- Stage 5: output projections ----
  cvt_pad_kernel<<<cdiv_sz((size_t)BB * APAD, 256), 256, 0, stream>>>(ctx, ctxb, BB, AA, APAD);
  make_bt_kernel<<<cdiv_sz((size_t)AA * APAD, 256), 256, 0, stream>>>(Wout, AA, 0, AA, AA, APAD, WoutT);
  gemm(ctxb, APAD, WoutT, APAD, ctx2, AA, BB, AA, APAD, bout);
  cvt_pad_kernel<<<cdiv_sz((size_t)BB * APAD, 256), 256, 0, stream>>>(ctx2, ctx2b, BB, AA, APAD);
  make_bt_kernel<<<cdiv_sz((size_t)DD * APAD, 256), 256, 0, stream>>>(Wo, DD, 0, AA, DD, APAD, WoT);
  gemm(ctx2b, APAD, WoT, APAD, out, DD, BB, DD, APAD, bo);
}
